// SSMBlock_62302795596627
// MI455X (gfx1250) — compile-verified, matrix operand order checked by reading, not run on hardware
//
#include <hip/hip_runtime.h>

typedef __attribute__((ext_vector_type(16))) __bf16 v16bf;
typedef __attribute__((ext_vector_type(8)))  float  v8f;
typedef __attribute__((ext_vector_type(4)))  unsigned int u32x4;
typedef __attribute__((ext_vector_type(8)))  int i32x8;
typedef __attribute__((ext_vector_type(4)))  int i32x4;

#define TOKENS   8192
#define DMODEL   1024
#define NSTATE   256
#define SEQLEN   2048
#define NBATCH   4
#define DFF_REAL 2730
#define DFF_PAD  2752
#define EPSV     1e-6f

#if defined(__has_builtin)
#if __has_builtin(__builtin_amdgcn_tensor_load_to_lds) && __has_builtin(__builtin_amdgcn_s_wait_tensorcnt)
#define HAVE_TDM 1
#endif
#endif
#ifndef HAVE_TDM
#define HAVE_TDM 0
#endif

union BFBits { unsigned short u; __bf16 b; };

static __device__ __forceinline__ unsigned short f32_to_bf16(float f){
  unsigned int u = __float_as_uint(f);
  unsigned int r = 0x7FFFu + ((u >> 16) & 1u);
  return (unsigned short)((u + r) >> 16);
}

static __device__ __forceinline__ __bf16 us_to_bf16(unsigned short u){
  BFBits t; t.u = u; return t.b;
}

static __device__ __forceinline__ void unpack8(v16bf &f, int base, uint4 d){
  unsigned int w[4] = {d.x, d.y, d.z, d.w};
#pragma unroll
  for (int i = 0; i < 4; ++i){
    f[base + 2*i + 0] = us_to_bf16((unsigned short)(w[i] & 0xFFFFu));
    f[base + 2*i + 1] = us_to_bf16((unsigned short)(w[i] >> 16));
  }
}

// One 16(row) x 32(K) bf16 fragment. `p` points at (row0, k0), ld = row stride
// in elements (multiple of 8 -> 16B aligned vector loads). CDNA5 16-bit A/B
// layout: lanes 0-15 -> K 0-7,16-23 ; lanes 16-31 -> K 8-15,24-31.
static __device__ __forceinline__ v16bf load_frag(const unsigned short* p, int ld){
  int lane = threadIdx.x & 31;
  int idx  = lane & 15;
  int hf   = lane >> 4;
  const unsigned short* r = p + (size_t)idx * ld + 8 * hf;
  uint4 d0 = *reinterpret_cast<const uint4*>(r);
  uint4 d1 = *reinterpret_cast<const uint4*>(r + 16);
  v16bf f;
  unpack8(f, 0, d0);
  unpack8(f, 8, d1);
  return f;
}

static __device__ __forceinline__ v8f wmma_bf16(v16bf a, v16bf b, v8f c){
  return __builtin_amdgcn_wmma_f32_16x16x32_bf16(false, a, false, b, (short)0, c, false, false);
}

static __device__ __forceinline__ v8f zero8(){
  v8f z = {0.f,0.f,0.f,0.f,0.f,0.f,0.f,0.f};
  return z;
}

static __device__ __forceinline__ float sigmoidf_fast(float x){
  return __builtin_amdgcn_rcpf(1.0f + __expf(-x));   // v_rcp_f32, avoids div chain
}

#if HAVE_TDM
// Issue one Tensor-Data-Mover 2D tile load: rows x row_elems bf16 elements,
// global row stride = stride_elems, into LDS at lds_addr (bytes). D# bit
// layout per CDNA5 ISA ch.8 (group0: count/lds/global/type, group1: data_size,
// tensor_dim0/1, tile_dim0/1, dim0 stride; groups 2/3 zero => 2D tensor).
// This toolchain's builtin takes 6 args (g0, g1, g2, g3, extra group, cpol).
static __device__ __forceinline__ void tdm_load_2d(unsigned lds_addr, const void* gptr,
                                                   unsigned rows, unsigned row_elems,
                                                   unsigned tensor_rows, unsigned stride_elems){
  unsigned long long ga = (unsigned long long)gptr;
  unsigned lo = __builtin_amdgcn_readfirstlane((unsigned)(ga & 0xFFFFFFFFu));
  unsigned hi = __builtin_amdgcn_readfirstlane((unsigned)(ga >> 32));
  u32x4 g0;
  g0[0] = 1u;                                        // count=1 user descriptor
  g0[1] = lds_addr;                                  // LDS byte address
  g0[2] = lo;                                        // global_addr[31:0]
  g0[3] = (hi & 0x1FFFFFFu) | 0x80000000u;           // global_addr[56:32] | type=2
  i32x8 g1;
  g1[0] = (int)(1u << 16);                           // data_size = 1 -> 2 bytes
  g1[1] = (int)((row_elems & 0xFFFFu) << 16);        // tensor_dim0[15:0]
  g1[2] = (int)(((row_elems >> 16) & 0xFFFFu) | ((tensor_rows & 0xFFFFu) << 16));
  g1[3] = (int)(((tensor_rows >> 16) & 0xFFFFu) | ((row_elems & 0xFFFFu) << 16)); // tile_dim0
  g1[4] = (int)(rows & 0xFFFFu);                     // tile_dim1 (tile_dim2=0)
  g1[5] = (int)stride_elems;                         // tensor_dim0_stride[31:0]
  g1[6] = 0;
  g1[7] = 0;
  i32x4 g2 = {0,0,0,0};
  i32x4 g3 = {0,0,0,0};
  i32x8 g4 = {0,0,0,0,0,0,0,0};
  __builtin_amdgcn_tensor_load_to_lds(g0, g1, g2, g3, g4, 0);
}
#endif

static __device__ __forceinline__ unsigned lds_offset_of(const void* p){
  return (unsigned)(unsigned long long)p;            // generic addr low 32 bits = LDS offset
}

// ---------------------------------------------------------------- converts
__global__ void k_cvt(const float* __restrict__ src, unsigned short* __restrict__ dst,
                      int ncopy, int ntotal){
  for (int i = blockIdx.x * blockDim.x + threadIdx.x; i < ntotal; i += gridDim.x * blockDim.x)
    dst[i] = (i < ncopy) ? f32_to_bf16(src[i]) : (unsigned short)0;
}

__global__ void k_cvt_w3(const float* __restrict__ src, unsigned short* __restrict__ dst){
  const int n = DMODEL * DFF_PAD;
  for (int i = blockIdx.x * blockDim.x + threadIdx.x; i < n; i += gridDim.x * blockDim.x){
    int row = i / DFF_PAD, col = i - row * DFF_PAD;
    dst[i] = (col < DFF_REAL) ? f32_to_bf16(src[row * DFF_REAL + col]) : (unsigned short)0;
  }
}

// ---------------------------------------------------------------- cayley
__global__ void k_cayley(const float* __restrict__ Ls, const float* __restrict__ Rs,
                         float* __restrict__ LQ, float* __restrict__ RQ){
  __shared__ float M[16][17];
  __shared__ float Rh[16][17];
  __shared__ float fac[16];
  __shared__ float piv;
  int blk = blockIdx.x;
  const float* S = (blk < 16) ? (Ls + blk * 256) : (Rs + (blk - 16) * 256);
  float* Q       = (blk < 16) ? (LQ + blk * 256) : (RQ + (blk - 16) * 256);
  int t = threadIdx.x, i = t >> 4, j = t & 15;
  float a = S[i*16 + j] - S[j*16 + i];
  M[i][j]  = (i == j ? 1.f : 0.f) + a;
  Rh[i][j] = (i == j ? 1.f : 0.f) - a;
  __syncthreads();
  for (int k = 0; k < 16; ++k){
    if (t == 0) piv = 1.f / M[k][k];
    __syncthreads();
    if (j == 0) fac[i] = (i == k) ? 0.f : M[i][k] * piv;
    __syncthreads();
    float f  = fac[i];
    float mk = M[k][j], rk = Rh[k][j];   // row k never modified (fac[k]==0)
    M[i][j]  -= f * mk;
    Rh[i][j] -= f * rk;
    __syncthreads();
  }
  Q[i*16 + j] = Rh[i][j] / M[i][i];
}

// ---------------------------------------------------------------- rmsnorm1 -> bf16 xs
__global__ void k_rmsnorm1(const float* __restrict__ x, const float* __restrict__ w,
                           unsigned short* __restrict__ xs){
  __shared__ float red[256];
  size_t token = blockIdx.x;
  const float* row = x + token * DMODEL;
  int t = threadIdx.x;
  float s = 0.f;
  for (int d = t; d < DMODEL; d += 256){ float v = row[d]; s += v * v; }
  red[t] = s; __syncthreads();
  for (int off = 128; off > 0; off >>= 1){ if (t < off) red[t] += red[t + off]; __syncthreads(); }
  float rinv = rsqrtf(red[0] / (float)DMODEL + EPSV);
  for (int d = t; d < DMODEL; d += 256)
    xs[token * DMODEL + d] = f32_to_bf16(w[d] * row[d] * rinv);
}

// ---------------------------------------------------------------- gates (WMMA, 2 waves)
__global__ void k_gates(const unsigned short* __restrict__ xs, const unsigned short* __restrict__ Wgb,
                        const float* __restrict__ bg,
                        float* __restrict__ alpha, float* __restrict__ beta){
  int mt = blockIdx.x;
  int wv = threadIdx.x >> 5;
  int lane = threadIdx.x & 31, idx = lane & 15, hf = lane >> 4;
  v8f acc = zero8();
  const unsigned short* A0 = xs  + (size_t)mt * 16 * DMODEL;
  const unsigned short* B0 = Wgb + (size_t)(wv * 16) * DMODEL;
  for (int k = 0; k < DMODEL; k += 32){
    v16bf a = load_frag(A0 + k, DMODEL);
    v16bf b = load_frag(B0 + k, DMODEL);
    acc = wmma_bf16(a, b, acc);
  }
  int col = wv * 16 + idx;
  float bgc = bg[col];
#pragma unroll
  for (int r = 0; r < 8; ++r){
    int m = r + 8 * hf;
    size_t token = (size_t)mt * 16 + m;
    float v = sigmoidf_fast(acc[r] + bgc);
    if (col < 16) alpha[token * 16 + col]      = v;
    else          beta [token * 16 + col - 16] = v;
  }
}

// ---------------------------------------------------------------- Bx = xs @ WB^T + bB (f32)
__global__ void k_gemm_bx(const unsigned short* __restrict__ xs, const unsigned short* __restrict__ WBb,
                          const float* __restrict__ bB, float* __restrict__ Bx){
  int mt = blockIdx.x;
  int wv = threadIdx.x >> 5;
  int lane = threadIdx.x & 31, idx = lane & 15, hf = lane >> 4;
  int col0 = blockIdx.y * 128 + wv * 16;
  v8f acc = zero8();
  const unsigned short* A0 = xs  + (size_t)mt * 16 * DMODEL;
  const unsigned short* B0 = WBb + (size_t)col0 * DMODEL;
  for (int k = 0; k < DMODEL; k += 32){
    v16bf a = load_frag(A0 + k, DMODEL);
    v16bf b = load_frag(B0 + k, DMODEL);
    acc = wmma_bf16(a, b, acc);
  }
  int col = col0 + idx;
  float bc = bB[col];
#pragma unroll
  for (int r = 0; r < 8; ++r){
    int m = r + 8 * hf;
    size_t token = (size_t)mt * 16 + m;
    Bx[token * NSTATE + col] = acc[r] + bc;
  }
}

// ---------------------------------------------------------------- sequential scan (per batch)
__global__ void k_scan(const float* __restrict__ LQ, const float* __restrict__ RQ,
                       const float* __restrict__ alpha, const float* __restrict__ beta,
                       const float* __restrict__ Bx, unsigned short* __restrict__ hs){
  __shared__ float lq[16][16][16];
  __shared__ float rq[16][16][16];
  __shared__ float h [16][16];
  __shared__ float hr[16][16];
  __shared__ float ab[32];
  int b = blockIdx.x;
  int t = threadIdx.x, i = t >> 4, c = t & 15;
  for (int n = t; n < 4096; n += 256){
    (&lq[0][0][0])[n] = LQ[n];
    (&rq[0][0][0])[n] = RQ[n];
  }
  h[i][c] = 0.f;
  __syncthreads();
  for (int s = 0; s < SEQLEN; ++s){
    size_t token = (size_t)b * SEQLEN + s;
    float abv = 0.f;
    if (t < 16)       abv = alpha[token * 16 + t];
    else if (t < 32)  abv = beta [token * 16 + (t - 16)];
    float bxv = Bx[token * NSTATE + t];
    __syncthreads();               // previous step's readers of ab are done
    if (t < 32) ab[t] = abv;
    __syncthreads();
    // stage 1: hr[j][c] = beta[j] * sum_k R_Q[j][c][k] * h[j][k]   (j == i here)
    float a1 = 0.f;
#pragma unroll
    for (int k = 0; k < 16; ++k) a1 += rq[i][c][k] * h[i][k];
    hr[i][c] = a1 * ab[16 + i];
    __syncthreads();
    // stage 2: h_new[i][c] = alpha[c] * sum_j L_Q[c][i][j] * hr[j][c] + bx
    float a2 = 0.f;
#pragma unroll
    for (int j = 0; j < 16; ++j) a2 += lq[c][i][j] * hr[j][c];
    float hn = a2 * ab[c] + bxv;
    hs[token * NSTATE + t] = f32_to_bf16(hn);
    h[i][c] = hn;
  }
}

// ---------------------------------------------------------------- y = hs@WC^T + bC ; xr = x+y ; rmsnorm2 -> xn
__global__ void k_resnorm(const unsigned short* __restrict__ hs, const unsigned short* __restrict__ WCb,
                          const float* __restrict__ bC, const float* __restrict__ x,
                          const float* __restrict__ n2w,
                          float* __restrict__ xres, unsigned short* __restrict__ xn){
  __shared__ float red[16];
  __shared__ float rinv[16];
  int mt = blockIdx.x;
  int wv = threadIdx.x >> 5;
  int lane = threadIdx.x & 31, idx = lane & 15, hf = lane >> 4;
  int t = threadIdx.x;
  if (t < 16) red[t] = 0.f;
  v8f acc[8];
#pragma unroll
  for (int tl = 0; tl < 8; ++tl) acc[tl] = zero8();
  const unsigned short* A0 = hs + (size_t)mt * 16 * NSTATE;
  for (int k = 0; k < NSTATE; k += 32){
    v16bf a = load_frag(A0 + k, NSTATE);           // reused across 8 tiles
#pragma unroll
    for (int tl = 0; tl < 8; ++tl){
      int col0 = tl * 128 + wv * 16;
      v16bf b = load_frag(WCb + (size_t)col0 * NSTATE + k, NSTATE);
      acc[tl] = wmma_bf16(a, b, acc[tl]);
    }
  }
  __syncthreads();
  float xr[8][8];
  float psum[8];
#pragma unroll
  for (int r = 0; r < 8; ++r) psum[r] = 0.f;
#pragma unroll
  for (int tl = 0; tl < 8; ++tl){
    int col = tl * 128 + wv * 16 + idx;
    float bc = bC[col];
#pragma unroll
    for (int r = 0; r < 8; ++r){
      int m = r + 8 * hf;
      size_t token = (size_t)mt * 16 + m;
      float v = acc[tl][r] + bc + x[token * DMODEL + col];
      xr[tl][r] = v;
      psum[r] += v * v;
    }
  }
#pragma unroll
  for (int r = 0; r < 8; ++r)
    atomicAdd(&red[r + 8 * hf], psum[r]);          // LDS ds_add_f32
  __syncthreads();
  if (t < 16) rinv[t] = rsqrtf(red[t] / (float)DMODEL + EPSV);
  __syncthreads();
#pragma unroll
  for (int tl = 0; tl < 8; ++tl){
    int col = tl * 128 + wv * 16 + idx;
    float wn = n2w[col];
#pragma unroll
    for (int r = 0; r < 8; ++r){
      int m = r + 8 * hf;
      size_t token = (size_t)mt * 16 + m;
      float v = xr[tl][r];
      xres[token * DMODEL + col] = v;
      xn  [token * DMODEL + col] = f32_to_bf16(wn * v * rinv[m]);
    }
  }
}

// ---------------------------------------------------------------- FFN1: g = silu(xn@w1^T) * (xn@w2^T)
// A tile (16 x DMODEL bf16, 32 KB) staged in LDS via TDM, shared by all waves.
__global__ void k_ffn1(const unsigned short* __restrict__ xn,
                       const unsigned short* __restrict__ w1b, const unsigned short* __restrict__ w2b,
                       unsigned short* __restrict__ g){
  __shared__ __align__(16) unsigned short As[16 * DMODEL];
  int mt = blockIdx.x;
  int wv = threadIdx.x >> 5;
  int lane = threadIdx.x & 31, idx = lane & 15, hf = lane >> 4;
  const unsigned short* A0g = xn + (size_t)mt * 16 * DMODEL;
#if HAVE_TDM
  if (wv == 0)
    tdm_load_2d(lds_offset_of(As), A0g, 16, DMODEL, TOKENS, DMODEL);
  __builtin_amdgcn_s_wait_tensorcnt(0);
#else
  for (int n = threadIdx.x * 8; n < 16 * DMODEL; n += blockDim.x * 8)
    *reinterpret_cast<uint4*>(As + n) = *reinterpret_cast<const uint4*>(A0g + n);
#endif
  __syncthreads();

  int f0 = (blockIdx.y * 4 + wv) * 16;            // 43*4 = 172 tiles = DFF_PAD/16
  v8f acc1 = zero8(), acc2 = zero8();
  const unsigned short* B1 = w1b + (size_t)f0 * DMODEL;
  const unsigned short* B2 = w2b + (size_t)f0 * DMODEL;
  for (int k = 0; k < DMODEL; k += 32){
    __builtin_prefetch((const void*)(B1 + k + 512), 0, 0);
    __builtin_prefetch((const void*)(B2 + k + 512), 0, 0);
    v16bf a  = load_frag(As + k, DMODEL);         // ds_load_b128
    v16bf b1 = load_frag(B1 + k, DMODEL);
    acc1 = wmma_bf16(a, b1, acc1);
    v16bf b2 = load_frag(B2 + k, DMODEL);
    acc2 = wmma_bf16(a, b2, acc2);
  }
  int f = f0 + idx;
#pragma unroll
  for (int r = 0; r < 8; ++r){
    int m = r + 8 * hf;
    size_t token = (size_t)mt * 16 + m;
    float u = acc1[r];
    float gv = (u * sigmoidf_fast(u)) * acc2[r];
    g[token * DFF_PAD + f] = f32_to_bf16(gv);
  }
}

// ---------------------------------------------------------------- FFN2 + final residual -> d_out
// A tile (16 x DFF_PAD bf16, 88 KB) staged in LDS via TDM; 2 output tiles/wave.
__global__ void k_ffn2(const unsigned short* __restrict__ g, const unsigned short* __restrict__ w3b,
                       const float* __restrict__ xres, float* __restrict__ out){
  __shared__ __align__(16) unsigned short As[16 * DFF_PAD];
  int mt = blockIdx.x;
  int wv = threadIdx.x >> 5;
  int lane = threadIdx.x & 31, idx = lane & 15, hf = lane >> 4;
  const unsigned short* A0g = g + (size_t)mt * 16 * DFF_PAD;
#if HAVE_TDM
  if (wv == 0)
    tdm_load_2d(lds_offset_of(As), A0g, 16, DFF_PAD, TOKENS, DFF_PAD);
  __builtin_amdgcn_s_wait_tensorcnt(0);
#else
  for (int n = threadIdx.x * 8; n < 16 * DFF_PAD; n += blockDim.x * 8)
    *reinterpret_cast<uint4*>(As + n) = *reinterpret_cast<const uint4*>(A0g + n);
#endif
  __syncthreads();

  int col0 = blockIdx.y * 256 + wv * 16;          // this wave: col0 and col0+128
  v8f acc0 = zero8(), acc1 = zero8();
  const unsigned short* B0 = w3b + (size_t)col0 * DFF_PAD;
  const unsigned short* B1 = w3b + (size_t)(col0 + 128) * DFF_PAD;
  for (int k = 0; k < DFF_PAD; k += 32){
    __builtin_prefetch((const void*)(B0 + k + 512), 0, 0);
    __builtin_prefetch((const void*)(B1 + k + 512), 0, 0);
    v16bf a  = load_frag(As + k, DFF_PAD);        // ds_load_b128
    v16bf b0 = load_frag(B0 + k, DFF_PAD);
    acc0 = wmma_bf16(a, b0, acc0);
    v16bf b1 = load_frag(B1 + k, DFF_PAD);
    acc1 = wmma_bf16(a, b1, acc1);
  }
#pragma unroll
  for (int r = 0; r < 8; ++r){
    int m = r + 8 * hf;
    size_t token = (size_t)mt * 16 + m;
    int ca = col0 + idx, cb = col0 + 128 + idx;
    out[token * DMODEL + ca] = xres[token * DMODEL + ca] + acc0[r];
    out[token * DMODEL + cb] = xres[token * DMODEL + cb] + acc1[r];
  }
}

// ---------------------------------------------------------------- launcher
extern "C" void kernel_launch(void* const* d_in, const int* in_sizes, int n_in,
                              void* d_out, int out_size, void* d_ws, size_t ws_size,
                              hipStream_t stream) {
  (void)in_sizes; (void)n_in; (void)out_size; (void)ws_size;
  const float* x      = (const float*)d_in[0];
  const float* L_skew = (const float*)d_in[1];
  const float* R_skew = (const float*)d_in[2];
  const float* Wg     = (const float*)d_in[3];
  const float* bg     = (const float*)d_in[4];
  const float* WB     = (const float*)d_in[5];
  const float* bB     = (const float*)d_in[6];
  const float* WC     = (const float*)d_in[7];
  const float* bC     = (const float*)d_in[8];
  const float* n1w    = (const float*)d_in[9];
  const float* n2w    = (const float*)d_in[10];
  const float* w1     = (const float*)d_in[11];
  const float* w2     = (const float*)d_in[12];
  const float* w3     = (const float*)d_in[13];
  float* out = (float*)d_out;

  char* ws = (char*)d_ws;
  size_t off = 0;
  auto alloc = [&](size_t bytes) -> void* {
    void* p = ws + off;
    off = (off + bytes + 255) & ~(size_t)255;
    return p;
  };
  float*          LQ    = (float*)alloc(4096 * sizeof(float));
  float*          RQ    = (float*)alloc(4096 * sizeof(float));
  unsigned short* Wgb   = (unsigned short*)alloc((size_t)32 * DMODEL * 2);
  unsigned short* WBb   = (unsigned short*)alloc((size_t)NSTATE * DMODEL * 2);
  unsigned short* WCb   = (unsigned short*)alloc((size_t)DMODEL * NSTATE * 2);
  unsigned short* w1b   = (unsigned short*)alloc((size_t)DFF_PAD * DMODEL * 2);
  unsigned short* w2b   = (unsigned short*)alloc((size_t)DFF_PAD * DMODEL * 2);
  unsigned short* w3b   = (unsigned short*)alloc((size_t)DMODEL * DFF_PAD * 2);
  unsigned short* xs    = (unsigned short*)alloc((size_t)TOKENS * DMODEL * 2);
  float*          alpha = (float*)alloc((size_t)TOKENS * 16 * sizeof(float));
  float*          beta  = (float*)alloc((size_t)TOKENS * 16 * sizeof(float));
  float*          Bx    = (float*)alloc((size_t)TOKENS * NSTATE * sizeof(float));
  unsigned short* hsb   = (unsigned short*)alloc((size_t)TOKENS * NSTATE * 2);
  float*          xres  = (float*)alloc((size_t)TOKENS * DMODEL * sizeof(float));
  unsigned short* xnb   = (unsigned short*)alloc((size_t)TOKENS * DMODEL * 2);
  unsigned short* gb    = (unsigned short*)alloc((size_t)TOKENS * DFF_PAD * 2);

  const int MT = TOKENS / 16;   // 512 token tiles

  // 1) weight conversions (bf16, padded)
  k_cvt<<<256, 256, 0, stream>>>(Wg, Wgb, 32 * DMODEL, 32 * DMODEL);
  k_cvt<<<512, 256, 0, stream>>>(WB, WBb, NSTATE * DMODEL, NSTATE * DMODEL);
  k_cvt<<<512, 256, 0, stream>>>(WC, WCb, DMODEL * NSTATE, DMODEL * NSTATE);
  k_cvt<<<1024, 256, 0, stream>>>(w1, w1b, DFF_REAL * DMODEL, DFF_PAD * DMODEL);
  k_cvt<<<1024, 256, 0, stream>>>(w2, w2b, DFF_REAL * DMODEL, DFF_PAD * DMODEL);
  k_cvt_w3<<<1024, 256, 0, stream>>>(w3, w3b);

  // 2) cayley orthogonal blocks
  k_cayley<<<32, 256, 0, stream>>>(L_skew, R_skew, LQ, RQ);

  // 3) norm1 -> xs ; gates ; Bx
  k_rmsnorm1<<<TOKENS, 256, 0, stream>>>(x, n1w, xs);
  k_gates<<<MT, 64, 0, stream>>>(xs, Wgb, bg, alpha, beta);
  k_gemm_bx<<<dim3(MT, 2), 256, 0, stream>>>(xs, WBb, bB, Bx);

  // 4) sequential scan
  k_scan<<<NBATCH, 256, 0, stream>>>(LQ, RQ, alpha, beta, Bx, hsb);

  // 5) y + residual + norm2
  k_resnorm<<<MT, 256, 0, stream>>>(hsb, WCb, bC, x, n2w, xres, xnb);

  // 6) FFN
  k_ffn1<<<dim3(MT, DFF_PAD / 64), 128, 0, stream>>>(xnb, w1b, w2b, gb);
  k_ffn2<<<dim3(MT, DMODEL / 256), 256, 0, stream>>>(gb, w3b, xres, out);
}
